// MockGQAAttention_33878702031220
// MI455X (gfx1250) — matrix operand order repaired and checked by reading, not compiled, not run on hardware
//
#include <hip/hip_runtime.h>
#include <cstdint>
#include <cstddef>

// ---------------------------------------------------------------- types ----
typedef __bf16 bf16;
typedef __attribute__((ext_vector_type(16))) __bf16 v16bf;
typedef __attribute__((ext_vector_type(8)))  float  v8f;
typedef __attribute__((ext_vector_type(4)))  unsigned int u32x4;
typedef __attribute__((ext_vector_type(4)))  int i32x4;
typedef __attribute__((ext_vector_type(8)))  int i32x8;

#define DEV static __device__ __forceinline__

constexpr int Bsz = 2, Sq = 2048, Hdim = 2048, NQh = 32, NKVh = 8, HD = 64, Gq = NQh / NKVh;

union FragU { v16bf v; uint4 q[2]; };

// load a 16-element bf16 fragment from two 16B-aligned addresses
DEV v16bf ld_frag2(const bf16* p0, const bf16* p1) {
  FragU f;
  f.q[0] = *reinterpret_cast<const uint4*>(p0);
  f.q[1] = *reinterpret_cast<const uint4*>(p1);
  return f.v;
}

DEV v8f wmma_bf16(v16bf a, v16bf b, v8f c) {
  // D = A(16x32 bf16) x B(32x16 bf16) + C(16x16 f32)
  return __builtin_amdgcn_wmma_f32_16x16x32_bf16(false, a, false, b, (short)0, c, false, false);
}

// CDNA5 async memory->LDS copy: per-lane 16B, tracked by ASYNCcnt.
DEV void async_ld_b128(uint32_t lds, const void* gaddr) {
  asm volatile("global_load_async_to_lds_b128 %0, %1, off"
               :: "v"(lds), "v"((uint64_t)(uintptr_t)gaddr) : "memory");
}
DEV void wait_async0() { asm volatile("s_wait_asynccnt 0x0" ::: "memory"); }

// ---------------------------------------------------------------- TDM ------
// One descriptor DMAs a 128-row x 32-elem bf16 tile (row stride K elems) from
// global into LDS, with hardware padding: after every 16 DWORDs (one 64B tile
// row) insert 4 DWORDs (16B) -> LDS row stride 80B == LDS_STRIDE bf16 elems.
#define HAS_TDM __has_builtin(__builtin_amdgcn_tensor_load_to_lds)

#if HAS_TDM
DEV void tdm_load_tile(uint32_t lds_addr, const void* gptr, uint32_t Kelems) {
  const uint64_t ga = (uint64_t)(uintptr_t)gptr;
  // D# group0: count=1 (valid), lds_addr, 57b global addr, type=2 ("image")
  u32x4 g0;
  g0[0] = 1u;
  g0[1] = lds_addr;
  g0[2] = (uint32_t)ga;
  g0[3] = (uint32_t)((ga >> 32) & 0x01FFFFFFu) | (2u << 30);
  // D# group1: data_size=1 (2B) | pad_enable | pad_interval=16dw | pad_amount=4dw
  const uint32_t dim0 = Kelems;   // tensor_dim0 (elems) for OOB
  const uint32_t dim1 = 128u;     // rows available from tile start
  i32x8 g1;
  g1[0] = (int)((1u << 16) | (1u << 20) | (3u << 22) | (3u << 25));
  g1[1] = (int)((dim0 & 0xFFFFu) << 16);                   // dim0[15:0]
  g1[2] = (int)((dim0 >> 16) | ((dim1 & 0xFFFFu) << 16));  // dim0[31:16], dim1[15:0]
  g1[3] = (int)((dim1 >> 16) | (32u << 16));               // dim1[31:16], tile_dim0=32
  g1[4] = (int)128;                                        // tile_dim1=128, tile_dim2=0
  g1[5] = (int)Kelems;                                     // tensor_dim0_stride[31:0]
  g1[6] = 0;                                               // stride hi / dim1_stride lo
  g1[7] = 0;
  const i32x4 z4 = {0, 0, 0, 0};
#if __clang_major__ >= 23
  const i32x8 z8 = {0, 0, 0, 0, 0, 0, 0, 0};
  __builtin_amdgcn_tensor_load_to_lds(g0, g1, z4, z4, z8, 0);
#else
  __builtin_amdgcn_tensor_load_to_lds(g0, g1, z4, z4, 0);
#endif
}
DEV void wait_tensor0() {
#if __has_builtin(__builtin_amdgcn_s_wait_tensorcnt)
  __builtin_amdgcn_s_wait_tensorcnt(0);
#else
  asm volatile("s_wait_tensorcnt 0x0" ::: "memory");
#endif
}
#endif

// lane-swizzle (stays within 16-lane halves for xor < 16)
#define SWZF(x, imm) __int_as_float(__builtin_amdgcn_ds_swizzle(__float_as_int(x), (imm)))

DEV float half_wave_max(float v) {
  v = fmaxf(v, SWZF(v, 0x041f)); // xor 1
  v = fmaxf(v, SWZF(v, 0x081f)); // xor 2
  v = fmaxf(v, SWZF(v, 0x101f)); // xor 4
  v = fmaxf(v, SWZF(v, 0x201f)); // xor 8
  return v;
}
DEV float half_wave_sum(float v) {
  v += SWZF(v, 0x041f);
  v += SWZF(v, 0x081f);
  v += SWZF(v, 0x101f);
  v += SWZF(v, 0x201f);
  return v;
}

// ------------------------------------------------------------- convert ----
__global__ void cvt_f32_to_bf16(const float* __restrict__ src, bf16* __restrict__ dst, int n) {
  for (int i = blockIdx.x * blockDim.x + threadIdx.x; i < n; i += gridDim.x * blockDim.x)
    dst[i] = (bf16)src[i];
}

// ----------------------------------------------------------------- GEMM ----
// C[M,N] = A[M,K] * W[N,K]^T + bias[N]
// mode 0: out bf16 head-major      [B, NH, S, 64]   (Q, K)
// mode 1: out bf16 head-transposed [B, NH, 64, S]   (V, for PV B-operand)
// mode 2: out f32 row-major        [M, N]           (final projection)
#define LDS_K 32
#define LDS_STRIDE 40  // bf16 elems; 80 bytes per row -> 16B-aligned frag loads

__global__ __launch_bounds__(128) void gemm_bf16_wmma(
    const bf16* __restrict__ A, const bf16* __restrict__ W,
    const float* __restrict__ bias, void* __restrict__ out,
    int M, int N, int K, int mode, int NH, int S)
{
  __shared__ __align__(16) bf16 AsT[2][128 * LDS_STRIDE];
  __shared__ __align__(16) bf16 BsT[2][128 * LDS_STRIDE];

  const int tid  = threadIdx.x;
  const int lane = tid & 31;
  const int wave = tid >> 5;
  const int wm = wave >> 1, wn = wave & 1;      // 2x2 wave grid, 64x64 per wave
  const int g = lane >> 4, r16 = lane & 15;
  const int mBase = blockIdx.y * 128;
  const int nBase = blockIdx.x * 128;

  v8f acc[4][4] = {};

#if HAS_TDM
  // one TDM descriptor per tile per buffer; wave0 -> A tile, wave1 -> B tile
  const uint32_t ldsA[2] = { (uint32_t)(uintptr_t)&AsT[0][0], (uint32_t)(uintptr_t)&AsT[1][0] };
  const uint32_t ldsB[2] = { (uint32_t)(uintptr_t)&BsT[0][0], (uint32_t)(uintptr_t)&BsT[1][0] };
  auto issue_tile = [&](int k0, int buf) {
    if (wave == 0)      tdm_load_tile(ldsA[buf], A + (size_t)mBase * K + k0, (uint32_t)K);
    else if (wave == 1) tdm_load_tile(ldsB[buf], W + (size_t)nBase * K + k0, (uint32_t)K);
  };
  auto wait_tiles = [&]() { if (wave < 2) wait_tensor0(); };
#else
  // fallback: per-lane async copies (ASYNCcnt)
  const bf16* gArow = A + (size_t)(mBase + tid) * K;
  const bf16* gWrow = W + (size_t)(nBase + tid) * K;
  const uint32_t ldsA0 = (uint32_t)(uintptr_t)&AsT[0][tid * LDS_STRIDE];
  const uint32_t ldsA1 = (uint32_t)(uintptr_t)&AsT[1][tid * LDS_STRIDE];
  const uint32_t ldsB0 = (uint32_t)(uintptr_t)&BsT[0][tid * LDS_STRIDE];
  const uint32_t ldsB1 = (uint32_t)(uintptr_t)&BsT[1][tid * LDS_STRIDE];
  auto issue_tile = [&](int k0, int buf) {
    const uint32_t la = buf ? ldsA1 : ldsA0;
    const uint32_t lb = buf ? ldsB1 : ldsB0;
#pragma unroll
    for (int q = 0; q < 4; q++) {
      async_ld_b128(la + q * 16, gArow + k0 + q * 8);
      async_ld_b128(lb + q * 16, gWrow + k0 + q * 8);
    }
  };
  auto wait_tiles = [&]() { wait_async0(); };
#endif

  issue_tile(0, 0);
  wait_tiles();
  __syncthreads();

  for (int k0 = 0; k0 < K; k0 += LDS_K) {
    const int cur = (k0 >> 5) & 1;
    if (k0 + LDS_K < K) issue_tile(k0 + LDS_K, cur ^ 1);  // DMA next tile

    v16bf af[4], bfr[4];
#pragma unroll
    for (int i = 0; i < 4; i++) {
      const bf16* p = &AsT[cur][(wm * 64 + i * 16 + r16) * LDS_STRIDE + g * 16];
      af[i] = ld_frag2(p, p + 8);               // A-frag: K = g*16 + e  (contiguous)
    }
#pragma unroll
    for (int j = 0; j < 4; j++) {
      const bf16* p = &BsT[cur][(wn * 64 + j * 16 + r16) * LDS_STRIDE + g * 16];
      bfr[j] = ld_frag2(p, p + 8);              // B-frag: K = g*16 + e  (contiguous)
    }
#pragma unroll
    for (int i = 0; i < 4; i++)
#pragma unroll
      for (int j = 0; j < 4; j++)
        acc[i][j] = wmma_bf16(af[i], bfr[j], acc[i][j]);

    wait_tiles();        // next tile landed in LDS
    __syncthreads();     // everyone done reading cur / writing nxt
  }

  // epilogue: C-frag row = base + rr + 8*g, col = base + (lane&15)
#pragma unroll
  for (int i = 0; i < 4; i++)
#pragma unroll
    for (int j = 0; j < 4; j++)
#pragma unroll
      for (int rr = 0; rr < 8; rr++) {
        int m = mBase + wm * 64 + i * 16 + g * 8 + rr;
        int n = nBase + wn * 64 + j * 16 + r16;
        float v = acc[i][j][rr] + bias[n];
        if (mode == 2) {
          reinterpret_cast<float*>(out)[(size_t)m * N + n] = v;
        } else {
          int bb = m / S, s = m - bb * S;
          int h = n >> 6, d = n & 63;
          size_t off = (mode == 0)
              ? ((((size_t)bb * NH + h) * S + s) << 6) + d
              : ((((size_t)bb * NH + h) << 6) + d) * (size_t)S + s;
          reinterpret_cast<bf16*>(out)[off] = (bf16)v;
        }
      }
}

// ---------------------------------------------------- flash attention ----
// Q head-major [B,NQ,S,64], K head-major [B,NKV,S,64], V transposed [B,NKV,64,S]
// out: bf16 [B,S,NQ*64]. One block = one (b, h, 64-row q tile); wave owns 16 rows.
__global__ __launch_bounds__(128) void gqa_flash_attn(
    const bf16* __restrict__ Qh, const bf16* __restrict__ Kh,
    const bf16* __restrict__ Vt, bf16* __restrict__ Oout)
{
  __shared__ __align__(16) bf16 Plds[4 * 16 * 64];  // per-wave P tile (16x64)

  const int tid = threadIdx.x, lane = tid & 31, w = tid >> 5;
  const int g = lane >> 4, c16 = lane & 15;
  const int b = blockIdx.z, h = blockIdx.y, kv = h / Gq;
  const int q0 = blockIdx.x * 64 + w * 16;

  const bf16* Qp = Qh + ((size_t)b * NQh  + h)  * Sq * HD;
  const bf16* Kp = Kh + ((size_t)b * NKVh + kv) * Sq * HD;
  const bf16* Vp = Vt + ((size_t)b * NKVh + kv) * HD * (size_t)Sq;

  // Q A-frags for 16 rows x 64 d, split into two K=32 chunks; reused all loop
  v16bf aq[2];
#pragma unroll
  for (int c = 0; c < 2; c++) {
    const bf16* p = Qp + (size_t)(q0 + c16) * HD + c * 32 + g * 16;
    aq[c] = ld_frag2(p, p + 8);
  }

  v8f oacc[4] = {};
  float mrun[8], lrun[8];
#pragma unroll
  for (int rr = 0; rr < 8; rr++) { mrun[rr] = -1e30f; lrun[rr] = 0.f; }
  const float SCL = 0.125f * 1.44269504088896340736f;  // d^-0.5 * log2(e)

  bf16* Pw = &Plds[w * 16 * 64];

  for (int kt = 0; kt < Sq; kt += 64) {
    // warm caches one tile ahead (global_prefetch_b8)
    if (kt + 64 < Sq) {
      __builtin_prefetch(Kp + (size_t)(kt + 64 + c16) * HD, 0, 0);
      __builtin_prefetch(Vp + (size_t)c16 * Sq + kt + 64, 0, 0);
    }

    // S = Q * K^T for 16 q-rows x 64 k-cols  (8 wmma)
    v8f sacc[4];
#pragma unroll
    for (int j = 0; j < 4; j++) {
      const bf16* pk = Kp + (size_t)(kt + j * 16 + c16) * HD;
      v16bf bk0 = ld_frag2(pk + g * 16,      pk + g * 16 + 8);
      v16bf bk1 = ld_frag2(pk + 32 + g * 16, pk + 32 + g * 16 + 8);
      v8f z = {};
      z = wmma_bf16(aq[0], bk0, z);
      sacc[j] = wmma_bf16(aq[1], bk1, z);
    }

    // online softmax: each lane holds rows rr+8g (col c16); row lives in a
    // 16-lane half -> ds_swizzle xor-reductions
    float mnew[8], alpha[8];
#pragma unroll
    for (int rr = 0; rr < 8; rr++) {
      float v = fmaxf(fmaxf(sacc[0][rr], sacc[1][rr]), fmaxf(sacc[2][rr], sacc[3][rr]));
      v = half_wave_max(v);
      mnew[rr]  = fmaxf(mrun[rr], v);
      alpha[rr] = exp2f((mrun[rr] - mnew[rr]) * SCL);
    }
#pragma unroll
    for (int j = 0; j < 4; j++)
#pragma unroll
      for (int rr = 0; rr < 8; rr++)
        sacc[j][rr] = exp2f((sacc[j][rr] - mnew[rr]) * SCL);
#pragma unroll
    for (int rr = 0; rr < 8; rr++) {
      float rs = half_wave_sum(sacc[0][rr] + sacc[1][rr] + sacc[2][rr] + sacc[3][rr]);
      lrun[rr] = lrun[rr] * alpha[rr] + rs;
      mrun[rr] = mnew[rr];
    }
#pragma unroll
    for (int jj = 0; jj < 4; jj++)
#pragma unroll
      for (int rr = 0; rr < 8; rr++)
        oacc[jj][rr] *= alpha[rr];

    // C-frag -> A-frag relayout of P through wave-private LDS (row-major 16x64)
#pragma unroll
    for (int j = 0; j < 4; j++)
#pragma unroll
      for (int rr = 0; rr < 8; rr++)
        Pw[(g * 8 + rr) * 64 + j * 16 + c16] = (bf16)sacc[j][rr];
    asm volatile("s_wait_dscnt 0x0" ::: "memory");  // LDS store->load turnaround

    v16bf ap[2];
#pragma unroll
    for (int c = 0; c < 2; c++) {
      const bf16* p = Pw + c16 * 64 + c * 32 + g * 16;
      ap[c] = ld_frag2(p, p + 8);
    }

    // O += P * V   (V pre-transposed: rows are d, cols are s -> contiguous B-frags)
#pragma unroll
    for (int jj = 0; jj < 4; jj++) {
      const bf16* pv = Vp + (size_t)(jj * 16 + c16) * Sq + kt;
      v16bf bv0 = ld_frag2(pv + g * 16,      pv + g * 16 + 8);
      v16bf bv1 = ld_frag2(pv + 32 + g * 16, pv + 32 + g * 16 + 8);
      oacc[jj] = wmma_bf16(ap[0], bv0, oacc[jj]);
      oacc[jj] = wmma_bf16(ap[1], bv1, oacc[jj]);
    }
  }

  // epilogue: normalize and write bf16 [B,S,NQ*64]
#pragma unroll
  for (int jj = 0; jj < 4; jj++)
#pragma unroll
    for (int rr = 0; rr < 8; rr++) {
      int srow = q0 + g * 8 + rr;
      int d = jj * 16 + c16;
      float v = oacc[jj][rr] / lrun[rr];
      Oout[(((size_t)b * Sq + srow) * NQh + h) * HD + d] = (bf16)v;
    }
}

// ------------------------------------------------------------- launcher ----
extern "C" void kernel_launch(void* const* d_in, const int* in_sizes, int n_in,
                              void* d_out, int out_size, void* d_ws, size_t ws_size,
                              hipStream_t stream)
{
  const float* x  = (const float*)d_in[0];
  const float* Wq = (const float*)d_in[1];
  const float* bq = (const float*)d_in[2];
  const float* Wk = (const float*)d_in[3];
  const float* bk = (const float*)d_in[4];
  const float* Wv = (const float*)d_in[5];
  const float* bv = (const float*)d_in[6];
  const float* Wo = (const float*)d_in[7];
  const float* bo = (const float*)d_in[8];

  constexpr size_t nX   = (size_t)Bsz * Sq * Hdim;       // 8.4M
  constexpr size_t nWq  = (size_t)Hdim * Hdim;           // 4.2M
  constexpr size_t nWkv = (size_t)NKVh * HD * Hdim;      // 1.05M
  constexpr size_t nQ   = nX;
  constexpr size_t nKV  = (size_t)Bsz * NKVh * Sq * HD;  // 2.1M

  bf16* p = (bf16*)d_ws;
  bf16* xb   = p; p += nX;
  bf16* wqb  = p; p += nWq;
  bf16* wkb  = p; p += nWkv;
  bf16* wvb  = p; p += nWkv;
  bf16* wob  = p; p += nWq;
  bf16* Qhd  = p; p += nQ;
  bf16* Khd  = p; p += nKV;
  bf16* Vtr  = p; p += nKV;
  bf16* attn = p; p += nX;

  // fp32 -> bf16
  cvt_f32_to_bf16<<<1024, 256, 0, stream>>>(x,  xb,  (int)nX);
  cvt_f32_to_bf16<<<1024, 256, 0, stream>>>(Wq, wqb, (int)nWq);
  cvt_f32_to_bf16<<<512,  256, 0, stream>>>(Wk, wkb, (int)nWkv);
  cvt_f32_to_bf16<<<512,  256, 0, stream>>>(Wv, wvb, (int)nWkv);
  cvt_f32_to_bf16<<<1024, 256, 0, stream>>>(Wo, wob, (int)nWq);

  const int M = Bsz * Sq;  // 4096
  dim3 gQ(Hdim / 128, M / 128);          // 16 x 32
  dim3 gKV((NKVh * HD) / 128, M / 128);  // 4 x 32

  gemm_bf16_wmma<<<gQ,  128, 0, stream>>>(xb, wqb, bq, Qhd, M, Hdim,      Hdim, 0, NQh,  Sq);
  gemm_bf16_wmma<<<gKV, 128, 0, stream>>>(xb, wkb, bk, Khd, M, NKVh * HD, Hdim, 0, NKVh, Sq);
  gemm_bf16_wmma<<<gKV, 128, 0, stream>>>(xb, wvb, bv, Vtr, M, NKVh * HD, Hdim, 1, NKVh, Sq);

  dim3 gA(Sq / 64, NQh, Bsz);            // 32 x 32 x 2
  gqa_flash_attn<<<gA, 128, 0, stream>>>(Qhd, Khd, Vtr, attn);

  gemm_bf16_wmma<<<gQ, 128, 0, stream>>>(attn, wob, bo, d_out, M, Hdim, Hdim, 2, NQh, Sq);
}
